// BotGAT_11012296147175
// MI455X (gfx1250) — matrix operand order repaired and checked by reading, not compiled, not run on hardware
//
#include <hip/hip_runtime.h>
#include <hip/hip_bf16.h>

typedef __attribute__((ext_vector_type(16))) _Float16 v16h;
typedef __attribute__((ext_vector_type(8)))  float    v8f;

#define HID 128

// ---------------- small helpers ----------------

__device__ inline float leakyf(float x, float s) { return x >= 0.f ? x : s * x; }

__device__ inline void atomicMaxF(float* addr, float val) {
    unsigned int* ua = (unsigned int*)addr;
    unsigned int old = __float_as_uint(*addr);
    while (__uint_as_float(old) < val) {
        unsigned int prev = atomicCAS(ua, old, __float_as_uint(val));
        if (prev == old) break;
        old = prev;
    }
}

// edge i in [0,E): src=ei[i], dst=ei[E+i]; i in [E,E+N): self loop
__device__ inline void edge_sd(const int* ei, long long E, long long i, int& s, int& d) {
    if (i < E) { s = ei[i]; d = ei[E + i]; }
    else       { s = d = (int)(i - E); }
}

// ---------------- kernels ----------------

// x = leaky(num_prop @ Wn + bn, 0.01)   [N,5]x[5,128]
__global__ void k_node_in(const float* __restrict__ np, const float* __restrict__ Wn,
                          const float* __restrict__ bn, float* __restrict__ out, int N) {
    long long t = (long long)blockIdx.x * blockDim.x + threadIdx.x;
    if (t >= (long long)N * HID) return;
    int n = (int)(t >> 7), c = (int)(t & 127);
    float acc = bn[c];
    #pragma unroll
    for (int k = 0; k < 5; ++k) acc += np[n * 5 + k] * Wn[k * HID + c];
    out[t] = leakyf(acc, 0.01f);
}

// pack weight [K=128][N=128] row-major f32 -> col-major f16 Bt[n][k]
__global__ void k_pack_w(const float* __restrict__ W, _Float16* __restrict__ Bt) {
    int t = blockIdx.x * blockDim.x + threadIdx.x;
    if (t >= HID * HID) return;
    int k = t >> 7, n = t & 127;
    Bt[n * HID + k] = (_Float16)W[k * HID + n];
}

// f32 -> f16 convert
__global__ void k_cvt16(const float* __restrict__ x, _Float16* __restrict__ h, long long cnt) {
    long long t = (long long)blockIdx.x * blockDim.x + threadIdx.x;
    if (t < cnt) h[t] = (_Float16)x[t];
}

// C[nrows,128] = A16[nrows,128] @ B(Bt col-major f16) (+bias) (+leaky 0.01)
// 8 waves/block, wave = 16 rows x 128 cols, v_wmma_f32_16x16x32_f16.
// B is staged in LDS once per block (32 KB of the 320 KB WGP LDS) and the
// WMMA stream is fed with ds_load_b128 pairs instead of global re-reads.
__global__ void __launch_bounds__(256) k_gemm128(const _Float16* __restrict__ A,
                                                 const _Float16* __restrict__ Bt,
                                                 const float* __restrict__ bias,
                                                 float* __restrict__ C, int nrows, int act) {
    __shared__ _Float16 Bs[HID * HID];   // 32 KB, col-major: Bs[n*128 + k]

    // cooperative fill: 2048 x uint4, 256 threads -> 8 b128 loads/stores each
    {
        const uint4* gsrc = (const uint4*)Bt;
        uint4* ldst = (uint4*)Bs;
        #pragma unroll
        for (int i = 0; i < 8; ++i)
            ldst[threadIdx.x + i * 256] = gsrc[threadIdx.x + i * 256];
    }
    __syncthreads();

    const int lane  = threadIdx.x & 31;
    const int wave  = threadIdx.x >> 5;
    const int m0    = blockIdx.x * 128 + wave * 16;
    const int khalf = lane >> 4;
    const int lid   = lane & 15;

    int row = m0 + lid;
    if (row > nrows - 1) row = nrows - 1;   // clamp OOB loads; stores guarded below

    union U { v16h v; uint4 q[2]; };

    // A fragments for the 4 K-blocks (K=32 each)
    U a[4];
    #pragma unroll
    for (int kb = 0; kb < 4; ++kb) {
        const _Float16* p = A + (size_t)row * HID + kb * 32 + khalf * 8;
        a[kb].q[0] = *(const uint4*)(p);        // K = kb*32 + khalf*8 + [0..8)
        a[kb].q[1] = *(const uint4*)(p + 16);   // K = kb*32 + 16 + khalf*8 + [0..8)
    }

    v8f zero = {};
    v8f acc[8];
    #pragma unroll
    for (int nt = 0; nt < 8; ++nt) acc[nt] = zero;

    #pragma unroll
    for (int nt = 0; nt < 8; ++nt) {
        const int ncol = nt * 16 + lid;
        #pragma unroll
        for (int kb = 0; kb < 4; ++kb) {
            U b;
            const _Float16* p = Bs + ncol * HID + kb * 32 + khalf * 8;
            b.q[0] = *(const uint4*)(p);        // K = kb*32 + khalf*8 + [0..8)
            b.q[1] = *(const uint4*)(p + 16);   // K = kb*32 + 16 + khalf*8 + [0..8)
            acc[nt] = __builtin_amdgcn_wmma_f32_16x16x32_f16(
                false, a[kb].v, false, b.v, (short)0, acc[nt], false, false);
        }
    }

    // C layout: VGPR r -> row m0 + r + 8*khalf ; col = n0 + lid
    #pragma unroll
    for (int nt = 0; nt < 8; ++nt) {
        const int col = nt * 16 + lid;
        #pragma unroll
        for (int r = 0; r < 8; ++r) {
            const int rowOut = m0 + khalf * 8 + r;
            if (rowOut < nrows) {
                float v = acc[nt][r];
                if (bias) v += bias[col];
                if (act)  v = leakyf(v, 0.01f);
                C[(size_t)rowOut * HID + col] = v;
            }
        }
    }
}

// per (node, head): attention coefficients + init max/denom
__global__ void k_att_coef(const float* __restrict__ hl, const float* __restrict__ asrc,
                           const float* __restrict__ adst, float* __restrict__ als,
                           float* __restrict__ ald, float* __restrict__ mmax,
                           float* __restrict__ den, int N, int H) {
    long long t = (long long)blockIdx.x * blockDim.x + threadIdx.x;
    if (t >= (long long)N * H) return;
    int n = (int)(t / H), h = (int)(t % H);
    int dh = HID / H;
    float s = 0.f, d = 0.f;
    const float* hp = hl + (size_t)n * HID + h * dh;
    for (int j = 0; j < dh; ++j) { float v = hp[j]; s += v * asrc[h * dh + j]; d += v * adst[h * dh + j]; }
    als[t] = s; ald[t] = d; mmax[t] = -3.0e38f; den[t] = 0.f;
}

// out[n,c] = bias[c]
__global__ void k_init_out(float* __restrict__ out, const float* __restrict__ bias, int N) {
    long long t = (long long)blockIdx.x * blockDim.x + threadIdx.x;
    if (t >= (long long)N * HID) return;
    out[t] = bias[t & 127];
}

__global__ void k_edge_max(const int* __restrict__ ei, long long E, long long Etot,
                           const float* __restrict__ als, const float* __restrict__ ald,
                           float* __restrict__ mmax, int H) {
    long long t = (long long)blockIdx.x * blockDim.x + threadIdx.x;
    if (t >= Etot * H) return;
    long long e = t / H; int h = (int)(t % H);
    int s, d; edge_sd(ei, E, e, s, d);
    float ev = leakyf(als[(long long)s * H + h] + ald[(long long)d * H + h], 0.2f);
    atomicMaxF(&mmax[(long long)d * H + h], ev);
}

__global__ void k_edge_sum(const int* __restrict__ ei, long long E, long long Etot,
                           const float* __restrict__ als, const float* __restrict__ ald,
                           const float* __restrict__ mmax, float* __restrict__ den, int H) {
    long long t = (long long)blockIdx.x * blockDim.x + threadIdx.x;
    if (t >= Etot * H) return;
    long long e = t / H; int h = (int)(t % H);
    int s, d; edge_sd(ei, E, e, s, d);
    float ev = leakyf(als[(long long)s * H + h] + ald[(long long)d * H + h], 0.2f);
    float ex = expf(ev - mmax[(long long)d * H + h]);
    atomicAdd(&den[(long long)d * H + h], ex);
}

// wave per edge; lane handles 4 features (float4 gather + 4 float atomic adds)
__global__ void k_edge_msg(const int* __restrict__ ei, long long E, long long Etot,
                           const float* __restrict__ als, const float* __restrict__ ald,
                           const float* __restrict__ mmax, const float* __restrict__ den,
                           const float* __restrict__ hl, float* __restrict__ out, int H) {
    long long w = ((long long)blockIdx.x * blockDim.x + threadIdx.x) >> 5;
    int lane = threadIdx.x & 31;
    if (w >= Etot) return;
    int s, d; edge_sd(ei, E, w, s, d);
    int f0 = lane * 4;
    int dh = HID / H;
    int h  = f0 / dh;
    float ev = leakyf(als[(long long)s * H + h] + ald[(long long)d * H + h], 0.2f);
    float alpha = expf(ev - mmax[(long long)d * H + h]) / (den[(long long)d * H + h] + 1e-16f);
    float4 hv = *(const float4*)(hl + (size_t)s * HID + f0);
    float* op = out + (size_t)d * HID + f0;
    atomicAdd(op + 0, hv.x * alpha);
    atomicAdd(op + 1, hv.y * alpha);
    atomicAdd(op + 2, hv.z * alpha);
    atomicAdd(op + 3, hv.w * alpha);
}

// d_out[n,2] = x @ Wo2 + bo2
__global__ void k_out_final(const float* __restrict__ x, const float* __restrict__ Wo2,
                            const float* __restrict__ bo2, float* __restrict__ out, int N) {
    long long t = (long long)blockIdx.x * blockDim.x + threadIdx.x;
    if (t >= (long long)N * 2) return;
    int n = (int)(t >> 1), j = (int)(t & 1);
    float acc = bo2[j];
    const float* xp = x + (size_t)n * HID;
    #pragma unroll 8
    for (int k = 0; k < HID; ++k) acc += xp[k] * Wo2[k * 2 + j];
    out[t] = acc;
}

// ---------------- launch ----------------

static inline unsigned cdiv_u(long long a, long long b) { return (unsigned)((a + b - 1) / b); }

extern "C" void kernel_launch(void* const* d_in, const int* in_sizes, int n_in,
                              void* d_out, int out_size, void* d_ws, size_t ws_size,
                              hipStream_t stream) {
    const float* num_prop = (const float*)d_in[0];
    const int*   ei       = (const int*)d_in[1];
    const float* Wn  = (const float*)d_in[2];
    const float* bn  = (const float*)d_in[3];
    const float* Wi  = (const float*)d_in[4];
    const float* bi  = (const float*)d_in[5];
    const float* W1  = (const float*)d_in[6];
    const float* a_src1 = (const float*)d_in[7];
    const float* a_dst1 = (const float*)d_in[8];
    const float* b1  = (const float*)d_in[9];
    const float* W2  = (const float*)d_in[10];
    const float* a_src2 = (const float*)d_in[11];
    const float* a_dst2 = (const float*)d_in[12];
    const float* b2  = (const float*)d_in[13];
    const float* Wo1 = (const float*)d_in[14];
    const float* bo1 = (const float*)d_in[15];
    const float* Wo2 = (const float*)d_in[16];
    const float* bo2 = (const float*)d_in[17];
    float* out = (float*)d_out;

    const int       N    = in_sizes[0] / 5;
    const long long E    = in_sizes[1] / 2;
    const long long Etot = E + (long long)N;
    const long long nfeat = (long long)N * HID;

    // workspace layout (256B aligned)
    auto al = [](size_t x) { return (x + 255) & ~(size_t)255; };
    char* ws = (char*)d_ws;
    size_t off = 0;
    float*    bufA = (float*)(ws + off);    off += al((size_t)nfeat * 4);
    float*    bufB = (float*)(ws + off);    off += al((size_t)nfeat * 4);
    _Float16* h16  = (_Float16*)(ws + off); off += al((size_t)nfeat * 2);
    _Float16* WiT  = (_Float16*)(ws + off); off += al((size_t)HID * HID * 2);
    _Float16* W1T  = (_Float16*)(ws + off); off += al((size_t)HID * HID * 2);
    _Float16* W2T  = (_Float16*)(ws + off); off += al((size_t)HID * HID * 2);
    _Float16* Wo1T = (_Float16*)(ws + off); off += al((size_t)HID * HID * 2);
    float*    als  = (float*)(ws + off);    off += al((size_t)N * 4 * 4);
    float*    ald  = (float*)(ws + off);    off += al((size_t)N * 4 * 4);
    float*    mmax = (float*)(ws + off);    off += al((size_t)N * 4 * 4);
    float*    den  = (float*)(ws + off);    off += al((size_t)N * 4 * 4);
    (void)ws_size;

    const int TB = 256;
    const unsigned gN128  = cdiv_u(nfeat, TB);
    const unsigned gGemm  = cdiv_u(N, 128);

    // weight packs (f32 -> f16 transposed)
    k_pack_w<<<cdiv_u(HID * HID, TB), TB, 0, stream>>>(Wi,  WiT);
    k_pack_w<<<cdiv_u(HID * HID, TB), TB, 0, stream>>>(W1,  W1T);
    k_pack_w<<<cdiv_u(HID * HID, TB), TB, 0, stream>>>(W2,  W2T);
    k_pack_w<<<cdiv_u(HID * HID, TB), TB, 0, stream>>>(Wo1, Wo1T);

    // input layer: bufA = leaky(np@Wn + bn)
    k_node_in<<<gN128, TB, 0, stream>>>(num_prop, Wn, bn, bufA, N);

    // bufB = leaky(bufA @ Wi + bi)
    k_cvt16<<<gN128, TB, 0, stream>>>(bufA, h16, nfeat);
    k_gemm128<<<gGemm, TB, 0, stream>>>(h16, WiT, bi, bufB, N, 1);

    // ---- GAT layer 1 (heads=4): hl = bufB @ W1 -> bufA; result -> bufB
    k_cvt16<<<gN128, TB, 0, stream>>>(bufB, h16, nfeat);
    k_gemm128<<<gGemm, TB, 0, stream>>>(h16, W1T, nullptr, bufA, N, 0);
    {
        const int H = 4;
        k_att_coef<<<cdiv_u((long long)N * H, TB), TB, 0, stream>>>(bufA, a_src1, a_dst1, als, ald, mmax, den, N, H);
        k_init_out<<<gN128, TB, 0, stream>>>(bufB, b1, N);
        k_edge_max<<<cdiv_u(Etot * H, TB), TB, 0, stream>>>(ei, E, Etot, als, ald, mmax, H);
        k_edge_sum<<<cdiv_u(Etot * H, TB), TB, 0, stream>>>(ei, E, Etot, als, ald, mmax, den, H);
        k_edge_msg<<<cdiv_u(Etot * 32, TB), TB, 0, stream>>>(ei, E, Etot, als, ald, mmax, den, bufA, bufB, H);
    }

    // ---- GAT layer 2 (heads=1): hl = bufB @ W2 -> bufA; result -> bufB
    k_cvt16<<<gN128, TB, 0, stream>>>(bufB, h16, nfeat);
    k_gemm128<<<gGemm, TB, 0, stream>>>(h16, W2T, nullptr, bufA, N, 0);
    {
        const int H = 1;
        k_att_coef<<<cdiv_u((long long)N * H, TB), TB, 0, stream>>>(bufA, a_src2, a_dst2, als, ald, mmax, den, N, H);
        k_init_out<<<gN128, TB, 0, stream>>>(bufB, b2, N);
        k_edge_max<<<cdiv_u(Etot * H, TB), TB, 0, stream>>>(ei, E, Etot, als, ald, mmax, H);
        k_edge_sum<<<cdiv_u(Etot * H, TB), TB, 0, stream>>>(ei, E, Etot, als, ald, mmax, den, H);
        k_edge_msg<<<cdiv_u(Etot * 32, TB), TB, 0, stream>>>(ei, E, Etot, als, ald, mmax, den, bufA, bufB, H);
    }

    // bufA = leaky(bufB @ Wo1 + bo1)
    k_cvt16<<<gN128, TB, 0, stream>>>(bufB, h16, nfeat);
    k_gemm128<<<gGemm, TB, 0, stream>>>(h16, Wo1T, bo1, bufA, N, 1);

    // out = bufA @ Wo2 + bo2
    k_out_final<<<cdiv_u((long long)N * 2, TB), TB, 0, stream>>>(bufA, Wo2, bo2, out, N);
}
